// Decoder_83708912599145
// MI455X (gfx1250) — matrix operand order repaired and checked by reading, not compiled, not run on hardware
//
#include <hip/hip_runtime.h>
#include <hip/hip_bf16.h>

// ---------------------------------------------------------------------------
// Pointer-generator decoder step for MI455X (gfx1250, wave32, WMMA bf16)
// B=64, S=400, ENC2=1024, DEC=512, EMB=256, V=50000
// Fragments are loaded as contiguous 32B chunks (fragment-major LDS layouts)
// so the inner loops are ds_load_b128 + v_wmma, not ds_load_u16 storms.
// ---------------------------------------------------------------------------

typedef __bf16 bf16_t;
typedef __attribute__((ext_vector_type(4)))  __bf16 v4bf;
typedef __attribute__((ext_vector_type(8)))  __bf16 v8bf;
typedef __attribute__((ext_vector_type(16))) __bf16 v16bf;
typedef __attribute__((ext_vector_type(8)))  float  v8f;

#define BDIM   64
#define S_LEN  400
#define ENC2   1024
#define DEC    512
#define EMB    256
#define VOCAB  50000
#define FEATD  1792   // DEC + ENC2 + EMB
#define XDIM   1280   // EMB + ENC2
#define MROWS  25600  // B*S

// ---------------------------------------------------------------------------
// small helper kernels
// ---------------------------------------------------------------------------
__global__ void k_embed(const int* __restrict__ input,
                        const float* __restrict__ emb_table,
                        float* __restrict__ emb) {
  int b = blockIdx.x, e = threadIdx.x;               // 64 x 256
  emb[b * EMB + e] = emb_table[(size_t)input[b] * EMB + e];
}

// We (1024x512) -> bf16, fragment-major swizzle:
//   We_sw[ ((kc*512 + n)*2 + h)*16 + i ] = We[kc*32 + 16h + i][n]
// so each 32-row k-chunk is a contiguous 32KB block, and a lane's whole
// B-fragment (16 bf16, K-ascending) is one aligned 32-byte read.
__global__ void k_swizzle_We(const float* __restrict__ attn_W,
                             bf16_t* __restrict__ We_sw) {
  const int kc = blockIdx.x >> 1;      // 0..31
  const int h  = blockIdx.x & 1;       // k half
  const int n  = threadIdx.x;          // 0..511
  v8bf lo, hi;
#pragma unroll
  for (int i = 0; i < 8; ++i)
    lo[i] = (bf16_t)attn_W[DEC * DEC + (kc * 32 + 16 * h + i) * DEC + n];
#pragma unroll
  for (int i = 0; i < 8; ++i)
    hi[i] = (bf16_t)attn_W[DEC * DEC + (kc * 32 + 16 * h + 8 + i) * DEC + n];
  bf16_t* dst = We_sw + ((size_t)(kc * DEC + n) * 2 + h) * 16;
  *(v8bf*)dst       = lo;
  *(v8bf*)(dst + 8) = hi;
}

// hW[b][n] = hidden[b] . Wh[:,n] + attn_b[n] + cov_b[n]
__global__ void k_hW(const float* __restrict__ hidden,
                     const float* __restrict__ attn_W,
                     const float* __restrict__ attn_b,
                     const float* __restrict__ cov_b,
                     float* __restrict__ hW) {
  int g = blockIdx.x * 256 + threadIdx.x;            // 64*512
  int b = g >> 9, n = g & 511;
  float acc = attn_b[n] + cov_b[n];
  const float* hb = hidden + b * DEC;
  for (int k = 0; k < DEC; ++k)
    acc += hb[k] * attn_W[k * DEC + n];
  hW[g] = acc;
}

// ---------------------------------------------------------------------------
// K2: fused attention:  score[b,s] = sum_n tanh( (enc@We)[r,n] + hW[b,n]
//                                  + cov[r]*cov_W[n] ) * v_w[n]
// WMMA bf16 16x16x32, block = 32 rows x 512 cols, 8 waves (2 Mtiles x 4 Nq)
// ---------------------------------------------------------------------------
__launch_bounds__(256)
__global__ void k_attn(const float* __restrict__ enc,
                       const bf16_t* __restrict__ We_sw,
                       const float* __restrict__ hW,
                       const float* __restrict__ covvec,
                       const float* __restrict__ cov_W,
                       const float* __restrict__ v_w,
                       float* __restrict__ score) {
  __shared__ __align__(32) bf16_t As[32 * 40];   // 32 rows, stride 40 (pad)
  __shared__ __align__(32) bf16_t Bs[32 * 512];  // fragment-major chunk copy
  __shared__ float sred[32];

  const int tid  = threadIdx.x;
  const int lane = tid & 31;
  const int w    = tid >> 5;
  const int mt   = w & 1;          // M tile within block (16 rows)
  const int nq   = w >> 1;         // 128-col quarter
  const int R0   = blockIdx.x * 32;
  const int l15  = lane & 15;
  const int h    = lane >> 4;

  v8f c[8] = {};                   // 8 N-tiles of 16 cols

  for (int kc = 0; kc < ENC2 / 32; ++kc) {
    __syncthreads();
    // stage A tile (32x32 f32 -> bf16), 4 elems/thread, one ds_store_b64
    {
      int e = tid * 4;
      int row = e >> 5, kk = e & 31;
      const float4 v =
          *(const float4*)(enc + (size_t)(R0 + row) * ENC2 + kc * 32 + kk);
      v4bf pk;
      pk[0] = (bf16_t)v.x; pk[1] = (bf16_t)v.y;
      pk[2] = (bf16_t)v.z; pk[3] = (bf16_t)v.w;
      *(v4bf*)(As + row * 40 + kk) = pk;
    }
    // stage B tile: contiguous 32KB copy of pre-swizzled We chunk
    {
      const uint4* src = (const uint4*)(We_sw + (size_t)kc * 32 * DEC);
      uint4* dst = (uint4*)Bs;
#pragma unroll
      for (int i = 0; i < 8; ++i) dst[tid * 8 + i] = src[tid * 8 + i];
    }
    __syncthreads();

    // A fragment: two aligned 16B LDS loads (K-ascending order)
    const bf16_t* ar = As + (16 * mt + l15) * 40;
    v8bf alo = *(const v8bf*)(ar + 16 * h);
    v8bf ahi = *(const v8bf*)(ar + 16 * h + 8);
    v16bf af = __builtin_shufflevector(alo, ahi, 0, 1, 2, 3, 4, 5, 6, 7,
                                       8, 9, 10, 11, 12, 13, 14, 15);
#pragma unroll
    for (int nt = 0; nt < 8; ++nt) {
      int n = 128 * nq + 16 * nt + l15;
      // whole B fragment: one aligned 32B LDS read
      v16bf bfr = *(const v16bf*)(Bs + ((n << 1) + h) * 16);
      c[nt] = __builtin_amdgcn_wmma_f32_16x16x32_bf16(
          false, af, false, bfr, (short)0, c[nt], false, false);
    }
  }

  // fused epilogue: tanh + dot with v_w, LDS reduce per row
  __syncthreads();
  if (tid < 32) sred[tid] = 0.0f;
  __syncthreads();

  const int rbase = R0 + 16 * mt;
  const int b = rbase / S_LEN;     // 400 % 16 == 0 -> tile within one batch
  float part[8], covv[8];
#pragma unroll
  for (int vi = 0; vi < 8; ++vi) {
    part[vi] = 0.0f;
    covv[vi] = covvec[rbase + vi + 8 * h];
  }
#pragma unroll
  for (int nt = 0; nt < 8; ++nt) {
    int n = 128 * nq + 16 * nt + l15;
    float hWn = hW[b * DEC + n];
    float cwn = cov_W[n];
    float vwn = v_w[n];
#pragma unroll
    for (int vi = 0; vi < 8; ++vi) {
      float e = c[nt][vi] + hWn + covv[vi] * cwn;
      part[vi] += tanhf(e) * vwn;
    }
  }
#pragma unroll
  for (int vi = 0; vi < 8; ++vi)
    atomicAdd(&sred[16 * mt + vi + 8 * h], part[vi]);   // ds_add_f32
  __syncthreads();
  if (tid < 32) score[R0 + tid] = sred[tid];
}

// ---------------------------------------------------------------------------
// K3: softmax over S + new_coverage + context = a @ enc   (one block per b)
// ---------------------------------------------------------------------------
__launch_bounds__(256)
__global__ void k_softmax_ctx(const float* __restrict__ score,
                              const float* __restrict__ covvec,
                              const float* __restrict__ enc,
                              float* __restrict__ a_out,
                              float* __restrict__ cov_out,
                              float* __restrict__ ctx) {
  __shared__ float red[256];
  __shared__ float aL[S_LEN];
  const int b = blockIdx.x, tid = threadIdx.x;
  const float* sc = score + b * S_LEN;

  float mx = -1e30f;
  for (int s = tid; s < S_LEN; s += 256) mx = fmaxf(mx, sc[s]);
  red[tid] = mx; __syncthreads();
  for (int o = 128; o > 0; o >>= 1) {
    if (tid < o) red[tid] = fmaxf(red[tid], red[tid + o]);
    __syncthreads();
  }
  mx = red[0]; __syncthreads();

  float sum = 0.0f;
  for (int s = tid; s < S_LEN; s += 256) {
    float e = expf(sc[s] - mx);
    aL[s] = e; sum += e;
  }
  red[tid] = sum; __syncthreads();
  for (int o = 128; o > 0; o >>= 1) {
    if (tid < o) red[tid] += red[tid + o];
    __syncthreads();
  }
  const float inv = 1.0f / red[0]; __syncthreads();

  for (int s = tid; s < S_LEN; s += 256) {
    float a = aL[s] * inv;
    aL[s] = a;
    a_out[b * S_LEN + s]   = a;
    cov_out[b * S_LEN + s] = covvec[b * S_LEN + s] + a;
  }
  __syncthreads();

  float acc[4] = {0.f, 0.f, 0.f, 0.f};
  const float* eb = enc + (size_t)b * S_LEN * ENC2;
  for (int s = 0; s < S_LEN; ++s) {
    float a = aL[s];
    const float* er = eb + (size_t)s * ENC2;
#pragma unroll
    for (int q = 0; q < 4; ++q) acc[q] += a * er[tid + 256 * q];
  }
#pragma unroll
  for (int q = 0; q < 4; ++q) ctx[b * ENC2 + tid + 256 * q] = acc[q];
}

// ---------------------------------------------------------------------------
// K4: GRU cell + assemble feat = [h_new, context, emb] (f32 + bf16)
// ---------------------------------------------------------------------------
__launch_bounds__(512)
__global__ void k_gru(const float* __restrict__ hidden,
                      const float* __restrict__ emb,
                      const float* __restrict__ ctx,
                      const float* __restrict__ Wih,
                      const float* __restrict__ Whh,
                      const float* __restrict__ bih,
                      const float* __restrict__ bhh,
                      float* __restrict__ hnew_out,
                      float* __restrict__ feat,
                      bf16_t* __restrict__ featb) {
  __shared__ float xL[XDIM];
  __shared__ float hL[DEC];
  const int b = blockIdx.x, tid = threadIdx.x;
  hL[tid] = hidden[b * DEC + tid];
  for (int i = tid; i < XDIM; i += 512)
    xL[i] = (i < EMB) ? emb[b * EMB + i] : ctx[b * ENC2 + (i - EMB)];
  __syncthreads();

  const int j = tid;
  const float* wr = Wih + (size_t)j * XDIM;
  const float* wz = Wih + (size_t)(DEC + j) * XDIM;
  const float* wn = Wih + (size_t)(2 * DEC + j) * XDIM;
  float ir = bih[j], iz = bih[DEC + j], inn = bih[2 * DEC + j];
  for (int k = 0; k < XDIM; ++k) {
    float x = xL[k];
    ir += x * wr[k]; iz += x * wz[k]; inn += x * wn[k];
  }
  const float* vr = Whh + (size_t)j * DEC;
  const float* vz = Whh + (size_t)(DEC + j) * DEC;
  const float* vn = Whh + (size_t)(2 * DEC + j) * DEC;
  float hr = bhh[j], hz = bhh[DEC + j], hn = bhh[2 * DEC + j];
  for (int k = 0; k < DEC; ++k) {
    float hh = hL[k];
    hr += hh * vr[k]; hz += hh * vz[k]; hn += hh * vn[k];
  }
  float r = 1.0f / (1.0f + expf(-(ir + hr)));
  float z = 1.0f / (1.0f + expf(-(iz + hz)));
  float n = tanhf(inn + r * hn);
  float hnew = (1.0f - z) * n + z * hL[j];

  hnew_out[b * DEC + j] = hnew;
  feat [(size_t)b * FEATD + j] = hnew;
  featb[(size_t)b * FEATD + j] = (bf16_t)hnew;
  for (int i = tid; i < XDIM; i += 512) {   // feat[512:1536]=ctx, [1536:]=emb
    float v = (i < ENC2) ? ctx[b * ENC2 + i] : emb[b * EMB + (i - ENC2)];
    feat [(size_t)b * FEATD + DEC + i] = v;
    featb[(size_t)b * FEATD + DEC + i] = (bf16_t)v;
  }
}

// K4b: p_gen = sigmoid(feat . pg_W + pg_b)
__global__ void k_pgen(const float* __restrict__ feat,
                       const float* __restrict__ pgW,
                       const float* __restrict__ pgb,
                       float* __restrict__ pgen, float* __restrict__ cscale) {
  __shared__ float red[256];
  const int b = blockIdx.x, tid = threadIdx.x;
  float s = 0.0f;
  for (int i = tid; i < FEATD; i += 256)
    s += feat[(size_t)b * FEATD + i] * pgW[i];
  red[tid] = s; __syncthreads();
  for (int o = 128; o > 0; o >>= 1) {
    if (tid < o) red[tid] += red[tid + o];
    __syncthreads();
  }
  if (tid == 0) {
    float p = 1.0f / (1.0f + expf(-(red[0] + pgb[0])));
    pgen[b] = p; cscale[b] = 1.0f - p;
  }
}

// ---------------------------------------------------------------------------
// K5a: prediction = feat_bf16 @ fc_W(+fc_b) via WMMA. One 16-col strip/block.
// fc_W (358 MB) streamed exactly once; B tile staged fragment-major so the
// inner loop is 1x ds_store_b64 (stage) + 1x 32B LDS read + wmma.
// ---------------------------------------------------------------------------
__launch_bounds__(128)
__global__ void k_fc(const bf16_t* __restrict__ featb,
                     const float* __restrict__ fcW,
                     const float* __restrict__ fcb,
                     float* __restrict__ pred) {
  __shared__ __align__(32) bf16_t Bs[32 * 16];  // fragment-major
  const int tid  = threadIdx.x;
  const int lane = tid & 31;
  const int w    = tid >> 5;          // wave -> M tile (16 rows)
  const int n0   = blockIdx.x * 16;
  const int l15  = lane & 15;
  const int h    = lane >> 4;
  // staging coords: thread = (column, k-quad)
  const int sn = tid & 15;
  const int kq = tid >> 4;            // 0..7
  const int sh = kq >> 2;             // k half of the quad

  v8f c = {};
  for (int kc = 0; kc < FEATD / 32; ++kc) {
    __syncthreads();
    {
      const float* src = fcW + (size_t)(kc * 32 + kq * 4) * VOCAB + n0 + sn;
      v4bf pk;
      pk[0] = (bf16_t)src[0];
      pk[1] = (bf16_t)src[VOCAB];
      pk[2] = (bf16_t)src[2 * (size_t)VOCAB];
      pk[3] = (bf16_t)src[3 * (size_t)VOCAB];
      *(v4bf*)(Bs + ((sn << 1) + sh) * 16 + (kq & 3) * 4) = pk;
    }
    __syncthreads();

    const bf16_t* ar = featb + (size_t)(16 * w + l15) * FEATD + kc * 32;
    v8bf alo = *(const v8bf*)(ar + 16 * h);
    v8bf ahi = *(const v8bf*)(ar + 16 * h + 8);
    v16bf af = __builtin_shufflevector(alo, ahi, 0, 1, 2, 3, 4, 5, 6, 7,
                                       8, 9, 10, 11, 12, 13, 14, 15);
    v16bf bfr = *(const v16bf*)(Bs + ((l15 << 1) + h) * 16);
    c = __builtin_amdgcn_wmma_f32_16x16x32_bf16(
        false, af, false, bfr, (short)0, c, false, false);
  }
  const int n = n0 + l15;
  const float bias = fcb[n];
#pragma unroll
  for (int vi = 0; vi < 8; ++vi) {
    int m = 16 * w + vi + 8 * h;
    pred[(size_t)m * VOCAB + n] = c[vi] + bias;
  }
}

// K5b: per-row softmax stats over V; vscale = p_gen / sum
__global__ void k_vstats(const float* __restrict__ pred,
                         const float* __restrict__ pgen,
                         float* __restrict__ rowmax, float* __restrict__ vscale) {
  __shared__ float red[256];
  const int b = blockIdx.x, tid = threadIdx.x;
  const float* pr = pred + (size_t)b * VOCAB;
  float mx = -1e30f;
  for (int v = tid; v < VOCAB; v += 256) mx = fmaxf(mx, pr[v]);
  red[tid] = mx; __syncthreads();
  for (int o = 128; o > 0; o >>= 1) {
    if (tid < o) red[tid] = fmaxf(red[tid], red[tid + o]);
    __syncthreads();
  }
  mx = red[0]; __syncthreads();
  float sum = 0.0f;
  for (int v = tid; v < VOCAB; v += 256) sum += expf(pr[v] - mx);
  red[tid] = sum; __syncthreads();
  for (int o = 128; o > 0; o >>= 1) {
    if (tid < o) red[tid] += red[tid + o];
    __syncthreads();
  }
  if (tid == 0) { rowmax[b] = mx; vscale[b] = pgen[b] / red[0]; }
}

// K5c: final_dist vocab part
__global__ void k_final(const float* __restrict__ pred,
                        const float* __restrict__ rowmax,
                        const float* __restrict__ vscale,
                        float* __restrict__ out) {
  size_t g = (size_t)blockIdx.x * 256 + threadIdx.x;
  int b = (int)(g / VOCAB);
  out[g] = expf(pred[g] - rowmax[b]) * vscale[b];
}

// K5d: scatter-add copy distribution
__global__ void k_scatter(const int* __restrict__ srcw,
                          const float* __restrict__ a_out,
                          const float* __restrict__ cscale,
                          float* __restrict__ fin) {
  const int b = blockIdx.x, tid = threadIdx.x;
  const float cs = cscale[b];
  for (int s = tid; s < S_LEN; s += 256) {
    int wv = srcw[b * S_LEN + s];
    atomicAdd(&fin[(size_t)b * VOCAB + wv], a_out[b * S_LEN + s] * cs);
  }
}

// ---------------------------------------------------------------------------
// host launcher
// ---------------------------------------------------------------------------
extern "C" void kernel_launch(void* const* d_in, const int* in_sizes, int n_in,
                              void* d_out, int out_size, void* d_ws, size_t ws_size,
                              hipStream_t stream) {
  (void)in_sizes; (void)n_in; (void)out_size; (void)ws_size;
  const int*   input   = (const int*)  d_in[0];
  const float* hidden  = (const float*)d_in[1];
  const float* enc     = (const float*)d_in[2];
  const int*   srcw    = (const int*)  d_in[3];
  const float* covvec  = (const float*)d_in[4];
  const float* emb_tab = (const float*)d_in[5];
  const float* attn_W  = (const float*)d_in[6];
  const float* attn_b  = (const float*)d_in[7];
  const float* v_w     = (const float*)d_in[8];
  const float* cov_W   = (const float*)d_in[9];
  const float* cov_b   = (const float*)d_in[10];
  const float* Wih     = (const float*)d_in[11];
  const float* Whh     = (const float*)d_in[12];
  const float* bih     = (const float*)d_in[13];
  const float* bhh     = (const float*)d_in[14];
  const float* fcW     = (const float*)d_in[15];
  const float* fcb     = (const float*)d_in[16];
  const float* pgW     = (const float*)d_in[17];
  const float* pgb     = (const float*)d_in[18];

  // workspace layout (bytes, 256-aligned sections)
  char* ws = (char*)d_ws;
  float*  emb     = (float*) (ws + 0);          //  64x256  f32   65536
  float*  hW      = (float*) (ws + 65536);      //  64x512  f32  131072
  bf16_t* We_sw   = (bf16_t*)(ws + 196608);     // 1024x512 bf16 1048576
  float*  score   = (float*) (ws + 1245184);    //  64x400  f32  102400
  float*  ctx     = (float*) (ws + 1347584);    //  64x1024 f32  262144
  float*  feat    = (float*) (ws + 1609728);    //  64x1792 f32  458752
  bf16_t* featb   = (bf16_t*)(ws + 2068480);    //  64x1792 bf16 229376
  float*  pgen    = (float*) (ws + 2297856);    //  64
  float*  cscale  = (float*) (ws + 2298112);    //  64
  float*  rowmax  = (float*) (ws + 2298368);    //  64
  float*  vscale  = (float*) (ws + 2298624);    //  64
  float*  pred    = (float*) (ws + 2298880);    //  64x50000 f32 12.8MB

  // output layout: [final_dist | h_new | a | new_coverage]
  float* out      = (float*)d_out;
  float* fin      = out;                    // 64*50000
  float* hnew_out = out + 3200000;          // 64*512
  float* a_out    = out + 3232768;          // 64*400
  float* cov_out  = out + 3258368;          // 64*400

  k_embed      <<<BDIM, 256, 0, stream>>>(input, emb_tab, emb);
  k_swizzle_We <<<64, 512, 0, stream>>>(attn_W, We_sw);
  k_hW         <<<(BDIM * DEC) / 256, 256, 0, stream>>>(hidden, attn_W, attn_b,
                                                        cov_b, hW);
  k_attn       <<<MROWS / 32, 256, 0, stream>>>(enc, We_sw, hW, covvec,
                                                cov_W, v_w, score);
  k_softmax_ctx<<<BDIM, 256, 0, stream>>>(score, covvec, enc,
                                          a_out, cov_out, ctx);
  k_gru        <<<BDIM, 512, 0, stream>>>(hidden, emb, ctx, Wih, Whh,
                                          bih, bhh, hnew_out, feat, featb);
  k_pgen       <<<BDIM, 256, 0, stream>>>(feat, pgW, pgb, pgen, cscale);
  k_fc         <<<VOCAB / 16, 128, 0, stream>>>(featb, fcW, fcb, pred);
  k_vstats     <<<BDIM, 256, 0, stream>>>(pred, pgen, rowmax, vscale);
  k_final      <<<(BDIM * VOCAB) / 256, 256, 0, stream>>>(pred, rowmax,
                                                          vscale, fin);
  k_scatter    <<<BDIM, 256, 0, stream>>>(srcw, a_out, cscale, fin);
}
